// PVConv_63960652972188
// MI455X (gfx1250) — matrix operand order repaired and checked by reading, not compiled, not run on hardware
//
#include <hip/hip_runtime.h>

typedef __attribute__((ext_vector_type(16))) _Float16 v16h;
typedef __attribute__((ext_vector_type(8)))  _Float16 v8h;
typedef __attribute__((ext_vector_type(8)))  float    v8f;

#define RR     32
#define RC     (RR * RR * RR)      /* 32768 */
#define BB     8
#define CC     64
#define NPTS   8192
#define BN_EPS 1e-4f

#define LDSROW 72                  /* padded ci row: 144B = 36 dwords, gcd(36,64)=4 */

union V16 { v16h v; v8h h[2]; };

/* ---------------- zero scratch (harness poisons ws with 0xAA) ------------- */
__global__ void pv_zero_f32(float* __restrict__ p, int n) {
    int i = blockIdx.x * blockDim.x + threadIdx.x;
    int stride = gridDim.x * blockDim.x;
    for (; i < n; i += stride) p[i] = 0.0f;
}

/* ---------------- coords normalization (one block per batch) -------------- */
__global__ void pv_normalize(const float* __restrict__ coords,
                             float* __restrict__ ncoords) {
    const int b = blockIdx.x;
    const int tid = threadIdx.x;
    __shared__ float red[256];
    __shared__ float meanv[3];
    __shared__ float sscale;
    const float* cb = coords + (size_t)b * 3 * NPTS;

    for (int d = 0; d < 3; ++d) {
        float s = 0.0f;
        for (int n = tid; n < NPTS; n += 256) s += cb[d * NPTS + n];
        red[tid] = s;
        __syncthreads();
        for (int off = 128; off > 0; off >>= 1) {
            if (tid < off) red[tid] += red[tid + off];
            __syncthreads();
        }
        if (tid == 0) meanv[d] = red[0] * (1.0f / NPTS);
        __syncthreads();
    }
    const float mx = meanv[0], my = meanv[1], mz = meanv[2];

    float m = 0.0f;
    for (int n = tid; n < NPTS; n += 256) {
        float x = cb[n] - mx;
        float y = cb[NPTS + n] - my;
        float z = cb[2 * NPTS + n] - mz;
        m = fmaxf(m, x * x + y * y + z * z);
    }
    red[tid] = m;
    __syncthreads();
    for (int off = 128; off > 0; off >>= 1) {
        if (tid < off) red[tid] = fmaxf(red[tid], red[tid + off]);
        __syncthreads();
    }
    if (tid == 0) sscale = fmaxf(sqrtf(red[0]) * 2.0f, 1e-20f);
    __syncthreads();

    const float inv = 1.0f / sscale;
    float* nb = ncoords + (size_t)b * 3 * NPTS;
    for (int d = 0; d < 3; ++d) {
        const float mean_d = meanv[d];
        for (int n = tid; n < NPTS; n += 256) {
            float v = ((cb[d * NPTS + n] - mean_d) * inv + 0.5f) * (float)RR;
            nb[d * NPTS + n] = fminf(fmaxf(v, 0.0f), (float)(RR - 1));
        }
    }
}

/* ---------------- convert conv weights [co][ci][27] -> f16 [t][co][ci] ---- */
__global__ void pv_wconvert(const float* __restrict__ w1,
                            const float* __restrict__ w2,
                            _Float16* __restrict__ wh1,
                            _Float16* __restrict__ wh2) {
    int id = blockIdx.x * blockDim.x + threadIdx.x;
    const int per = 27 * CC * CC;               /* 110592 */
    if (id >= 2 * per) return;
    const int which = id / per;
    const int r = id % per;
    const int t  = r >> 12;                     /* r / 4096 */
    const int co = (r >> 6) & 63;
    const int ci = r & 63;
    const float* src = which ? w2 : w1;
    _Float16*   dst = which ? wh2 : wh1;
    dst[r] = (_Float16)src[((co << 6) + ci) * 27 + t];
}

/* ---------------- scatter-average voxelization ---------------------------- */
__global__ void pv_scatter(const float* __restrict__ feats,
                           const float* __restrict__ ncoords,
                           float* __restrict__ gsum,
                           float* __restrict__ cnt) {
    int id = blockIdx.x * blockDim.x + threadIdx.x;
    if (id >= BB * NPTS) return;
    const int b = id / NPTS;
    const int n = id % NPTS;
    const float* nb = ncoords + (size_t)b * 3 * NPTS;
    const int vx = __float2int_rn(nb[n]);
    const int vy = __float2int_rn(nb[NPTS + n]);
    const int vz = __float2int_rn(nb[2 * NPTS + n]);
    const int idx = (vx * RR + vy) * RR + vz;
    atomicAdd(&cnt[(size_t)b * RC + idx], 1.0f);
    const float* fb = feats + (size_t)b * CC * NPTS + n;
    float* gb = gsum + (size_t)b * CC * RC + idx;
    for (int c = 0; c < CC; ++c)
        atomicAdd(gb + (size_t)c * RC, fb[(size_t)c * NPTS]);
}

/* ---------------- divide by counts, convert grid to f16 ------------------- */
__global__ void pv_finalize(const float* __restrict__ gsum,
                            const float* __restrict__ cnt,
                            _Float16* __restrict__ gh) {
    int id = blockIdx.x * blockDim.x + threadIdx.x;
    if (id >= BB * CC * RC) return;
    const int b = id / (CC * RC);
    const int v = id & (RC - 1);
    const float c = fmaxf(cnt[(size_t)b * RC + v], 1.0f);
    gh[id] = (_Float16)(gsum[id] / c);
}

/* ---------------- 3x3x3 conv + BN + leaky ReLU via WMMA implicit GEMM -----
 * One workgroup per (b, x, y): 8 wave32 produce 64 Cout x 32 z outputs.
 * Halo staged in LDS transposed to [row][z][ci] (ci innermost, padded to 72)
 * so every 16-bit B 32x16 fragment is two aligned ds_load_b128.
 * Weights double-buffered per tap in LDS [2][co][72] (one barrier per tap,
 * global weight loads overlap the WMMA stream). A 16x32 fragments are two
 * 16B LDS loads. 54 x v_wmma_f32_16x16x32_f16 per wave, fp32 accumulation.
 */
__global__ void __launch_bounds__(256)
pv_conv3d_wmma(const _Float16* __restrict__ gin,
               const _Float16* __restrict__ wh,
               const float* __restrict__ bias,
               const float* __restrict__ gamma,
               const float* __restrict__ beta,
               const float* __restrict__ mean,
               const float* __restrict__ var,
               _Float16* __restrict__ gout) {
    __shared__ __align__(16) _Float16 lds_in[9 * 34 * LDSROW];   /* 44064 B */
    __shared__ __align__(16) _Float16 lds_w[2][CC * LDSROW];     /* 18432 B */

    const int bid = blockIdx.x;
    const int b = bid >> 10;
    const int x = (bid >> 5) & 31;
    const int y = bid & 31;
    const int tid = threadIdx.x;

    /* stage input halo: 64 ci x 9 (dx,dy) x 34 z, zero padded; transpose so
     * ci is innermost in LDS. Global side stays z-coalesced. */
    const _Float16* gb = gin + (size_t)b * CC * RC;
    for (int e = tid; e < CC * 9 * 34; e += 256) {
        const int ci  = e / 306;
        const int r2  = e % 306;
        const int row = r2 / 34;           /* (dx+1)*3 + (dy+1) */
        const int zz  = r2 % 34;           /* padded z index */
        const int gx = x + row / 3 - 1;
        const int gy = y + row % 3 - 1;
        const int z  = zz - 1;
        _Float16 val = (_Float16)0.0f;
        if ((unsigned)gx < 32u && (unsigned)gy < 32u && (unsigned)z < 32u)
            val = gb[(size_t)ci * RC + (gx * RR + gy) * RR + z];
        lds_in[(row * 34 + zz) * LDSROW + ci] = val;
    }

    /* stage tap-0 weights into buffer 0 (16B vector copies) */
    {
        const _Float16* wt = wh;
#pragma unroll
        for (int i = tid; i < 512; i += 256) {
            const int co = i >> 3;
            const int c8 = (i & 7) * 8;
            *(v8h*)&lds_w[0][co * LDSROW + c8] = *(const v8h*)&wt[co * 64 + c8];
        }
    }
    __syncthreads();                       /* halo + tap-0 weights ready */

    const int wave  = tid >> 5;
    const int lane  = tid & 31;
    const int mtile = wave >> 1;           /* 0..3 : Cout group of 16 */
    const int ntile = wave & 1;            /* 0..1 : z group of 16   */
    const int mrow  = lane & 15;
    const int hi    = lane >> 4;           /* lane half */
    const int co_a  = mtile * 16 + mrow;   /* A-matrix M for this lane */
    const int zn    = ntile * 16 + mrow;   /* B-matrix N (output z)    */

    v8f acc = {};
    for (int t = 0; t < 27; ++t) {
        /* prefetch tap t+1 weights into the other buffer (no barrier needed:
         * buffer (t+1)&1 is not read this iteration) */
        if (t + 1 < 27) {
            const _Float16* wt = wh + (t + 1) * 4096;
            const int nb = (t + 1) & 1;
#pragma unroll
            for (int i = tid; i < 512; i += 256) {
                const int co = i >> 3;
                const int c8 = (i & 7) * 8;
                *(v8h*)&lds_w[nb][co * LDSROW + c8] = *(const v8h*)&wt[co * 64 + c8];
            }
        }

        const int row = t / 3;             /* t = row*3 + dz */
        const int dz  = t % 3;
        const _Float16* ap = &lds_w[t & 1][co_a * LDSROW];
        const _Float16* bp = &lds_in[(row * 34 + zn + dz) * LDSROW];
#pragma unroll
        for (int kc = 0; kc < 2; ++kc) {
            const int kb = kc * 32;
            V16 a, bm;
            /* A 16x32 layout: lane has K = kb+hi*8+{0..7} and kb+16+hi*8+{0..7} */
            a.h[0] = *(const v8h*)&ap[kb + hi * 8];
            a.h[1] = *(const v8h*)&ap[kb + 16 + hi * 8];
            /* B 32x16 layout: lane has K = kb+hi*8+{0..7} and kb+16+hi*8+{0..7} */
            bm.h[0] = *(const v8h*)&bp[kb + hi * 8];
            bm.h[1] = *(const v8h*)&bp[kb + 16 + hi * 8];
            acc = __builtin_amdgcn_wmma_f32_16x16x32_f16(
                false, a.v, false, bm.v, (short)0, acc, false, false);
        }
        /* publish next weight buffer / protect current one */
        __syncthreads();
    }

    /* epilogue: bias + eval-BN + leaky_relu(0.1), store f16.
     * f32 C/D 16x16 layout: lane l, VGPR v -> M = (l>=16?8:0)+v, N = l&15 */
    _Float16* ob = gout + (size_t)b * CC * RC + (x * RR + y) * RR;
#pragma unroll
    for (int v = 0; v < 8; ++v) {
        const int co = mtile * 16 + hi * 8 + v;
        const float invs = gamma[co] * rsqrtf(var[co] + BN_EPS);
        float val = (acc[v] + bias[co]) * invs + (beta[co] - mean[co] * invs);
        val = val > 0.0f ? val : 0.1f * val;
        ob[(size_t)co * RC + zn] = (_Float16)val;
    }
}

/* ---------------- trilinear devoxelize + point MLP branch + sum ----------- */
__global__ void pv_devox_point(const _Float16* __restrict__ grid,
                               const float* __restrict__ ncoords,
                               const float* __restrict__ feats,
                               const float* __restrict__ mw,
                               const float* __restrict__ mb,
                               const float* __restrict__ pg,
                               const float* __restrict__ pb,
                               const float* __restrict__ pm,
                               const float* __restrict__ pv,
                               float* __restrict__ out) {
    int id = blockIdx.x * blockDim.x + threadIdx.x;
    if (id >= BB * CC * NPTS) return;
    const int n  = id % NPTS;
    const int co = (id / NPTS) % CC;
    const int b  = id / (NPTS * CC);

    const float* nb = ncoords + (size_t)b * 3 * NPTS;
    const float cx = nb[n], cy = nb[NPTS + n], cz = nb[2 * NPTS + n];
    const int lx = (int)floorf(cx), ly = (int)floorf(cy), lz = (int)floorf(cz);
    const float fx = cx - (float)lx, fy = cy - (float)ly, fz = cz - (float)lz;
    const int hx = min(lx + 1, RR - 1);
    const int hy = min(ly + 1, RR - 1);
    const int hz = min(lz + 1, RR - 1);

    const _Float16* gb = grid + ((size_t)b * CC + co) * RC;
    float acc = 0.0f;
#pragma unroll
    for (int d = 0; d < 8; ++d) {
        const int ix = (d & 4) ? hx : lx;
        const int iy = (d & 2) ? hy : ly;
        const int iz = (d & 1) ? hz : lz;
        const float w = ((d & 4) ? fx : 1.0f - fx) *
                        ((d & 2) ? fy : 1.0f - fy) *
                        ((d & 1) ? fz : 1.0f - fz);
        acc += w * (float)gb[(ix * RR + iy) * RR + iz];
    }

    /* point branch: 1x1 conv + BN + relu (fp32, bandwidth-trivial) */
    const float* fb = feats + (size_t)b * CC * NPTS + n;
    const float* wr = mw + co * CC;
    float p = 0.0f;
    for (int ci = 0; ci < CC; ++ci)
        p = fmaf(wr[ci], fb[(size_t)ci * NPTS], p);
    const float invs = pg[co] * rsqrtf(pv[co] + BN_EPS);
    p = (p + mb[co]) * invs + (pb[co] - pm[co] * invs);
    p = fmaxf(p, 0.0f);

    out[id] = acc + p;
}

/* ---------------- coords passthrough (second tuple output) ---------------- */
__global__ void pv_copy(const float* __restrict__ src, float* __restrict__ dst, int n) {
    int i = blockIdx.x * blockDim.x + threadIdx.x;
    int stride = gridDim.x * blockDim.x;
    for (; i < n; i += stride) dst[i] = src[i];
}

extern "C" void kernel_launch(void* const* d_in, const int* in_sizes, int n_in,
                              void* d_out, int out_size, void* d_ws, size_t ws_size,
                              hipStream_t stream) {
    const float* feats  = (const float*)d_in[0];
    const float* coords = (const float*)d_in[1];
    const float* c1w = (const float*)d_in[2];
    const float* c1b = (const float*)d_in[3];
    const float* g1  = (const float*)d_in[4];
    const float* b1  = (const float*)d_in[5];
    const float* m1  = (const float*)d_in[6];
    const float* v1  = (const float*)d_in[7];
    const float* c2w = (const float*)d_in[8];
    const float* c2b = (const float*)d_in[9];
    const float* g2  = (const float*)d_in[10];
    const float* b2  = (const float*)d_in[11];
    const float* m2  = (const float*)d_in[12];
    const float* v2  = (const float*)d_in[13];
    const float* mw  = (const float*)d_in[14];
    const float* mb  = (const float*)d_in[15];
    const float* pg  = (const float*)d_in[16];
    const float* pb  = (const float*)d_in[17];
    const float* pm  = (const float*)d_in[18];
    const float* pv  = (const float*)d_in[19];

    /* workspace layout (bytes):
     * [0, 786432)            norm_coords f32 [B,3,N]
     * [786432, 1835008)      counts f32 [B, R^3]
     * [1835008, 2056192)     conv1 weights f16 [27][64][64]
     * [2056192, 2277376)     conv2 weights f16 [27][64][64]
     * [4194304, 71303168)    gsum f32 [B,64,R^3]  (later reused: gh1 f16 | gh2 f16)
     * [71303168, 104857600)  gh0 f16 [B,64,R^3]
     */
    char* ws = (char*)d_ws;
    float*    ncoords = (float*)(ws + 0);
    float*    cnt     = (float*)(ws + 786432);
    _Float16* wh1     = (_Float16*)(ws + 1835008);
    _Float16* wh2     = (_Float16*)(ws + 2056192);
    float*    gsum    = (float*)(ws + 4194304);
    _Float16* gh1     = (_Float16*)(ws + 4194304);            /* reuse after finalize */
    _Float16* gh2     = (_Float16*)(ws + 4194304 + 33554432);
    _Float16* gh0     = (_Float16*)(ws + 71303168);

    pv_zero_f32<<<2048, 256, 0, stream>>>(cnt, BB * RC);
    pv_zero_f32<<<8192, 256, 0, stream>>>(gsum, BB * CC * RC);
    pv_normalize<<<BB, 256, 0, stream>>>(coords, ncoords);
    pv_wconvert<<<(2 * 27 * CC * CC + 255) / 256, 256, 0, stream>>>(c1w, c2w, wh1, wh2);
    pv_scatter<<<(BB * NPTS + 255) / 256, 256, 0, stream>>>(feats, ncoords, gsum, cnt);
    pv_finalize<<<(BB * CC * RC + 255) / 256, 256, 0, stream>>>(gsum, cnt, gh0);

    pv_conv3d_wmma<<<BB * RR * RR, 256, 0, stream>>>(gh0, wh1, c1b, g1, b1, m1, v1, gh1);
    pv_conv3d_wmma<<<BB * RR * RR, 256, 0, stream>>>(gh1, wh2, c2b, g2, b2, m2, v2, gh2);

    pv_devox_point<<<(BB * CC * NPTS + 255) / 256, 256, 0, stream>>>(
        gh2, ncoords, feats, mw, mb, pg, pb, pm, pv, (float*)d_out);
    pv_copy<<<(BB * 3 * NPTS + 255) / 256, 256, 0, stream>>>(
        coords, (float*)d_out + (size_t)BB * CC * NPTS, BB * 3 * NPTS);
}
